// ScaledDotProductAttentionWithParaTopic_42013370089971
// MI455X (gfx1250) — matrix operand order repaired and checked by reading, not compiled, not run on hardware
//
#include <hip/hip_runtime.h>
#include <hip/hip_bf16.h>
#include <math.h>

typedef __attribute__((ext_vector_type(16))) __bf16 v16bf;
typedef __attribute__((ext_vector_type(8)))  __bf16 v8bf;
typedef __attribute__((ext_vector_type(4)))  __bf16 v4bf;
typedef __attribute__((ext_vector_type(8)))  float  v8f;

#define Bn 8
#define Hn 16
#define Ln 1024
#define Dn 64
#define DM 1024   // Hn*Dn

static __device__ inline v16bf combine16(v8bf lo, v8bf hi) {
  return __builtin_shufflevector(lo, hi, 0, 1, 2, 3, 4, 5, 6, 7,
                                 8, 9, 10, 11, 12, 13, 14, 15);
}

// ---------------- kernel 0: f32 -> bf16 pre-conversion (scale folded in) ----------------
__global__ __launch_bounds__(256) void cvt_kernel(
    const float* __restrict__ src, __bf16* __restrict__ dst, float scale)
{
  const size_t i = (size_t)blockIdx.x * 256 + threadIdx.x;   // one float4 per thread
  const float4 f = ((const float4*)src)[i];
  v4bf o;
  o[0] = (__bf16)(f.x * scale);
  o[1] = (__bf16)(f.y * scale);
  o[2] = (__bf16)(f.z * scale);
  o[3] = (__bf16)(f.w * scale);
  ((v4bf*)dst)[i] = o;
}

// ---------------- kernel 1: per-key topic gate MLP ----------------
__global__ __launch_bounds__(256) void gate_kernel(
    const float* __restrict__ pt, const float* __restrict__ W1,
    const float* __restrict__ b1, const float* __restrict__ W2,
    const float* __restrict__ b2, float* __restrict__ gate)
{
  __shared__ float sW1[64 * 64];
  __shared__ float sB1[64];
  __shared__ float sW2[64];
  const int tid = threadIdx.x;
  for (int i = tid; i < 64 * 64; i += 256) sW1[i] = W1[i];
  if (tid < 64) { sB1[tid] = b1[tid]; sW2[tid] = W2[tid]; }
  __syncthreads();

  const size_t idx = (size_t)blockIdx.x * 256 + tid;   // key index over B*H*L
  const float* x = pt + idx * 64;
  float xr[64];
  #pragma unroll
  for (int d = 0; d < 64; ++d) xr[d] = x[d];

  float g = b2[0];
  for (int j = 0; j < 64; ++j) {
    float acc = sB1[j];
    #pragma unroll
    for (int d = 0; d < 64; ++d) acc = fmaf(sW1[j * 64 + d], xr[d], acc);
    g = fmaf(sW2[j], tanhf(acc), g);
  }
  gate[idx] = 1.0f / (1.0f + __expf(-g));
}

// ---------------- kernel 2: fused attention ----------------
// One workgroup per (b, h, 16-row q tile). 8 waves; wave w owns key columns
// [w*128, w*128+128). Scores stay in WMMA accumulators; probs go to LDS (bf16)
// for the P@V pass; V tiles staged per-wave in LDS; cross-wave PV reduction
// via ds_add_f32.
__global__ __launch_bounds__(256) void attn_kernel(
    const __bf16* __restrict__ qbf, const __bf16* __restrict__ kbf,
    const __bf16* __restrict__ vbf, const float* __restrict__ bias,
    const float* __restrict__ gate,
    float* __restrict__ wout, __bf16* __restrict__ Ows)
{
  __shared__ __bf16 sP[16][1024];    // 32 KB: gated probs (bf16)
  __shared__ __bf16 sV[8][32][64];   // 32 KB: per-wave V tile (bf16)
  __shared__ float  sAcc[16][64];    //  4 KB: PV accumulator
  __shared__ float  sRmax[16][8];
  __shared__ float  sRsum[16][8];

  const int tid  = threadIdx.x;
  const int w    = tid >> 5;          // wave id 0..7
  const int lane = tid & 31;
  const int ln15 = lane & 15;
  const int hi   = (lane >> 4) << 3;  // 0 or 8: C/D row offset for upper lanes
  const int kob  = (lane < 16) ? 0 : 16;

  const int bid = blockIdx.x;
  const int qt  = bid & 63;
  const int h   = (bid >> 6) & 15;
  const int b   = bid >> 10;
  const int q0  = qt * 16;

  const size_t bh   = (size_t)b * Hn + h;
  const __bf16* qhh = qbf  + bh * Ln * Dn;
  const __bf16* khh = kbf  + bh * Ln * Dn;
  const __bf16* vhh = vbf  + bh * Ln * Dn;
  const float*  bbh = bias + (bh * Ln + q0) * Ln;
  const float*  gbh = gate + bh * Ln;

  // zero PV accumulator (visible after first __syncthreads)
  #pragma unroll
  for (int i = 0; i < 4; ++i) {
    int e = tid + i * 256;
    sAcc[e >> 6][e & 63] = 0.0f;
  }

  // ---- S = (q*scale) @ K^T + bias, via bf16 WMMA (scale pre-folded into qbf) ----
  const __bf16* qrow = qhh + (size_t)(q0 + ln15) * Dn;
  v16bf aF[2];
  #pragma unroll
  for (int dc = 0; dc < 2; ++dc) {
    const int s1 = dc * 32 + kob;
    aF[dc] = combine16(*(const v8bf*)(qrow + s1), *(const v8bf*)(qrow + s1 + 8));
  }

  v8f c[8] = {};
  #pragma unroll
  for (int cb = 0; cb < 8; ++cb) {
    const int c0 = w * 128 + cb * 16;
    const __bf16* krow = khh + (size_t)(c0 + ln15) * Dn;   // key row = B column
    #pragma unroll
    for (int dc = 0; dc < 2; ++dc) {
      const int s = dc * 32 + ((lane < 16) ? 0 : 16);
      const v16bf bF = *(const v16bf*)(krow + s);          // 32B aligned
      c[cb] = __builtin_amdgcn_wmma_f32_16x16x32_bf16(
          false, aF[dc], false, bF, (short)0, c[cb], false, false);
    }
    #pragma unroll
    for (int r = 0; r < 8; ++r)   // add bias in-register (single HBM read)
      c[cb][r] += bbh[(size_t)(r + hi) * Ln + (c0 + ln15)];
  }

  // ---- row-wise softmax: shfl within 16-lane halves + LDS cross-wave ----
  float M[8], S[8];
  #pragma unroll
  for (int r = 0; r < 8; ++r) {
    float m = c[0][r];
    #pragma unroll
    for (int cb = 1; cb < 8; ++cb) m = fmaxf(m, c[cb][r]);
    #pragma unroll
    for (int d = 1; d < 16; d <<= 1) m = fmaxf(m, __shfl_xor(m, d, 32));
    M[r] = m;
  }
  if (ln15 == 0) {
    #pragma unroll
    for (int r = 0; r < 8; ++r) sRmax[r + hi][w] = M[r];
  }
  __syncthreads();
  #pragma unroll
  for (int r = 0; r < 8; ++r) {
    float m = sRmax[r + hi][0];
    #pragma unroll
    for (int ww = 1; ww < 8; ++ww) m = fmaxf(m, sRmax[r + hi][ww]);
    M[r] = m;
  }
  #pragma unroll
  for (int r = 0; r < 8; ++r) {
    float s = 0.0f;
    #pragma unroll
    for (int cb = 0; cb < 8; ++cb) {
      float e = __expf(c[cb][r] - M[r]);
      c[cb][r] = e;
      s += e;
    }
    #pragma unroll
    for (int d = 1; d < 16; d <<= 1) s += __shfl_xor(s, d, 32);
    S[r] = s;
  }
  if (ln15 == 0) {
    #pragma unroll
    for (int r = 0; r < 8; ++r) sRsum[r + hi][w] = S[r];
  }
  __syncthreads();
  #pragma unroll
  for (int r = 0; r < 8; ++r) {
    float s = 0.0f;
    #pragma unroll
    for (int ww = 0; ww < 8; ++ww) s += sRsum[r + hi][ww];
    S[r] = 1.0f / s;
  }

  // ---- normalize * gate; single write of weights; stage bf16 probs in LDS ----
  float* wo = wout + (bh * Ln + q0) * Ln;
  #pragma unroll
  for (int cb = 0; cb < 8; ++cb) {
    const int col = w * 128 + cb * 16 + ln15;
    const float gv = gbh[col];
    #pragma unroll
    for (int r = 0; r < 8; ++r) {
      const int row = r + hi;
      const float p = c[cb][r] * S[r] * gv;
      wo[(size_t)row * Ln + col] = p;
      sP[row][col] = (__bf16)p;
    }
  }
  __syncthreads();

  // ---- O_partial = P[:, wave cols] @ V[wave cols, :], bf16 WMMA ----
  v8f o[4] = {};
  for (int kc = 0; kc < 4; ++kc) {
    const int kbase = w * 128 + kc * 32;
    // stage wave's 32-key V tile into its private LDS slice (coalesced 128B rows)
    {
      const __bf16* vrow = vhh + (size_t)(kbase + lane) * Dn;
      #pragma unroll
      for (int j = 0; j < 8; ++j)
        *(v8bf*)&sV[w][lane][j * 8] = *(const v8bf*)(vrow + j * 8);
    }
    __syncthreads();   // uniform across workgroup (4 iterations for all waves)

    v16bf aP;
    {
      const int s1 = kbase + kob;
      aP = combine16(*(const v8bf*)&sP[ln15][s1], *(const v8bf*)&sP[ln15][s1 + 8]);
    }
    const int ksl = (lane < 16) ? 0 : 16;
    #pragma unroll
    for (int cbv = 0; cbv < 4; ++cbv) {
      const int dcol = cbv * 16 + ln15;
      v16bf bV;
      #pragma unroll
      for (int e = 0; e < 16; ++e) bV[e] = sV[w][ksl + e][dcol];
      o[cbv] = __builtin_amdgcn_wmma_f32_16x16x32_bf16(
          false, aP, false, bV, (short)0, o[cbv], false, false);
    }
  }
  // cross-wave reduction via ds_add_f32
  #pragma unroll
  for (int cbv = 0; cbv < 4; ++cbv) {
    #pragma unroll
    for (int r = 0; r < 8; ++r)
      atomicAdd(&sAcc[r + hi][cbv * 16 + ln15], o[cbv][r]);
  }
  __syncthreads();

  __bf16* obase = Ows + (bh * Ln + q0) * Dn;
  #pragma unroll
  for (int i = 0; i < 4; ++i) {
    const int e = tid + i * 256;
    obase[e] = (__bf16)sAcc[e >> 6][e & 63];
  }
}

// ---------------- kernel 3: Y = X @ W_out^T + b_out (bf16 WMMA GEMM) ----------------
// X[m][kk] with m = b*1024 + l, kk = h*64 + dd  ->  Ows[((b*H+h)*L + l)*64 + dd]
__global__ __launch_bounds__(256) void proj_kernel(
    const __bf16* __restrict__ Xbf, const __bf16* __restrict__ Wbf,
    const float* __restrict__ bout, float* __restrict__ Y)
{
  const int tid  = threadIdx.x;
  const int wid  = tid >> 5;
  const int lane = tid & 31;
  const int ln15 = lane & 15;
  const int hi   = (lane >> 4) << 3;
  const int kob  = (lane < 16) ? 0 : 16;
  const int M0 = blockIdx.x * 64;
  const int N0 = blockIdx.y * 64;

  v8f acc[2] = {};
  for (int kb = 0; kb < DM; kb += 32) {
    #pragma unroll
    for (int i = 0; i < 2; ++i) {
      const int bi = wid * 2 + i;
      const int br = bi >> 2, bc = bi & 3;
      const int m  = M0 + br * 16 + ln15;
      const int bB = m >> 10, l = m & 1023;
      // both 8-element runs live in head (kb>>6); dd runs are contiguous bf16
      const __bf16* Xrow =
          Xbf + (((size_t)bB * Hn + (kb >> 6)) * Ln + l) * Dn + (kb & 63);
      const v16bf a =
          combine16(*(const v8bf*)(Xrow + kob), *(const v8bf*)(Xrow + 8 + kob));

      const int n  = N0 + bc * 16 + ln15;
      const int ks = kb + ((lane < 16) ? 0 : 16);
      const __bf16* Wrow = Wbf + (size_t)n * DM;
      __builtin_prefetch(Wrow + ks + 256, 0, 1);   // global_prefetch_b8 ahead of stream
      const v16bf bfr = *(const v16bf*)(Wrow + ks);
      acc[i] = __builtin_amdgcn_wmma_f32_16x16x32_bf16(
          false, a, false, bfr, (short)0, acc[i], false, false);
    }
  }
  #pragma unroll
  for (int i = 0; i < 2; ++i) {
    const int bi = wid * 2 + i;
    const int br = bi >> 2, bc = bi & 3;
    #pragma unroll
    for (int r = 0; r < 8; ++r) {
      const int row = M0 + br * 16 + r + hi;
      const int col = N0 + bc * 16 + ln15;
      Y[(size_t)row * DM + col] = acc[i][r] + bout[col];
    }
  }
}

extern "C" void kernel_launch(void* const* d_in, const int* in_sizes, int n_in,
                              void* d_out, int out_size, void* d_ws, size_t ws_size,
                              hipStream_t stream) {
  const float* q    = (const float*)d_in[0];
  const float* k    = (const float*)d_in[1];
  const float* v    = (const float*)d_in[2];
  const float* pt   = (const float*)d_in[3];
  const float* bias = (const float*)d_in[4];
  const float* W1   = (const float*)d_in[5];
  const float* b1   = (const float*)d_in[6];
  const float* W2   = (const float*)d_in[7];
  const float* b2   = (const float*)d_in[8];
  const float* Wo   = (const float*)d_in[9];
  const float* bo   = (const float*)d_in[10];

  float* out       = (float*)d_out;
  float* graph_out = out;                           // [B, L, DM]
  float* weights   = out + (size_t)Bn * Ln * DM;    // [B, H, L, L]

  // workspace layout (byte offsets, all 256B aligned)
  char* ws = (char*)d_ws;
  const size_t nqkv = (size_t)Bn * Hn * Ln * Dn;    // 8,388,608 elements
  float*  gate   = (float*)(ws);                                  // 512 KB
  __bf16* Owsbf  = (__bf16*)(ws + 524288);                        // 16 MB
  __bf16* qbf    = (__bf16*)(ws + 524288 + nqkv * 2);             // 16 MB
  __bf16* kbf    = (__bf16*)(ws + 524288 + nqkv * 4);             // 16 MB
  __bf16* vbf    = (__bf16*)(ws + 524288 + nqkv * 6);             // 16 MB
  __bf16* woutbf = (__bf16*)(ws + 524288 + nqkv * 8);             //  2 MB

  // pre-convert to bf16 (scale folded into q)
  cvt_kernel<<<(int)(nqkv / 1024), 256, 0, stream>>>(q, qbf, 0.125f);
  cvt_kernel<<<(int)(nqkv / 1024), 256, 0, stream>>>(k, kbf, 1.0f);
  cvt_kernel<<<(int)(nqkv / 1024), 256, 0, stream>>>(v, vbf, 1.0f);
  cvt_kernel<<<(DM * DM) / 1024, 256, 0, stream>>>(Wo, woutbf, 1.0f);

  gate_kernel<<<(Bn * Hn * Ln) / 256, 256, 0, stream>>>(pt, W1, b1, W2, b2, gate);
  attn_kernel<<<Bn * Hn * (Ln / 16), 256, 0, stream>>>(qbf, kbf, vbf, bias, gate,
                                                       weights, Owsbf);
  dim3 pg(Bn * Ln / 64, DM / 64);
  proj_kernel<<<pg, 256, 0, stream>>>(Owsbf, woutbf, bo, graph_out);
}
